// Mamba2_13846974562809
// MI455X (gfx1250) — matrix operand order repaired and checked by reading, not compiled, not run on hardware
//
#include <hip/hip_runtime.h>
#include <math.h>

// ---------------- model constants ----------------
#define D_MODEL   2048
#define D_STATE   128
#define D_CONV    4
#define HEADDIM   64
#define D_INNER   4096
#define D_SSM     4096
#define NHEADS    64
#define CONV_DIM  4352           // D_SSM + 2*D_STATE
#define D_IN_PROJ 8512           // 2*D_INNER + 2*D_STATE + NHEADS
#define BATCH     2
#define SEQLEN    2048
#define EPS       1e-5f

// ---------------- WMMA / TDM types ----------------
typedef __attribute__((ext_vector_type(16))) __bf16        v16bf;
typedef __attribute__((ext_vector_type(8)))  float         v8f;
typedef __attribute__((ext_vector_type(4)))  unsigned int  v4u;
typedef __attribute__((ext_vector_type(8)))  int           v8i;
typedef __attribute__((ext_vector_type(4)))  int           v4i;

#if defined(__has_builtin)
#if __has_builtin(__builtin_amdgcn_tensor_load_to_lds) && \
    __has_builtin(__builtin_amdgcn_s_wait_tensorcnt)
#define USE_TDM 1
#endif
#endif

union FragBF {
  v16bf v;
  uint4 u4[2];
  unsigned short s[16];
};

__device__ __forceinline__ unsigned short f32_to_bf16(float f) {
  unsigned int u = __float_as_uint(f);
  unsigned int r = u + 0x7FFFu + ((u >> 16) & 1u);   // round-to-nearest-even
  return (unsigned short)(r >> 16);
}

__device__ __forceinline__ float sigmoidf_fast(float x) {
  return 1.0f / (1.0f + __expf(-x));
}

// ---------------- fp32 -> bf16 conversion ----------------
__global__ __launch_bounds__(256)
void cvt_bf16_kernel(const float* __restrict__ in, unsigned short* __restrict__ out, size_t n) {
  size_t i = (size_t)blockIdx.x * blockDim.x + threadIdx.x;
  size_t stride = (size_t)gridDim.x * blockDim.x;
  for (; i < n; i += stride) out[i] = f32_to_bf16(in[i]);
}

// ---------------- WMMA GEMM: C(MxN) = A(MxK) * B(NxK)^T ----------------
// A, B bf16 row-major (K contiguous), C fp32 row-major.
// Block tile 128x64, 8 waves (4x2), each wave: 2x2 tiles of 16x16.
// A block tile (128 x 32) is DMA'd into LDS by the Tensor Data Mover with
// 16-byte padding every 64-byte row (LDS row stride = 80 B = 40 bf16).
#define TILE_M 128
#define TILE_N 64
#define KSTEP  32
#define A_LDS  40   // padded stride in bf16 elements

__global__ __launch_bounds__(256)
void wmma_gemm_bf16_kernel(const unsigned short* __restrict__ A,
                           const unsigned short* __restrict__ B,
                           float* __restrict__ C,
                           int M, int N, int K) {
  __shared__ unsigned short aTile[TILE_M * A_LDS];

  const int tid  = threadIdx.x;
  const int lane = tid & 31;
  const int wave = tid >> 5;            // 0..7
  const int wm   = (wave & 3) * 32;     // wave M offset in block tile
  const int wn   = (wave >> 2) * 32;    // wave N offset in block tile

  const int mBase = blockIdx.y * TILE_M;
  const int nBase = blockIdx.x * TILE_N;

  const int half16 = (lane < 16) ? 0 : 1;   // which K-half this lane holds
  const int rlane  = lane & 15;

  v8f acc[2][2];
#pragma unroll
  for (int i = 0; i < 2; ++i)
#pragma unroll
    for (int j = 0; j < 2; ++j) acc[i][j] = (v8f)0.0f;

#ifdef USE_TDM
  // ---- invariant pieces of the Tensor DMA Descriptor (D#) ----
  const unsigned ldsAddr = (unsigned)(size_t)(&aTile[0]);  // generic LDS addr truncates to LDS offset
  v8i g1;
  g1[0] = (int)((1u << 16)        // data_size = 1 -> 2 bytes
              | (1u << 20)        // pad_enable (LDS destination padding)
              | (3u << 22)        // pad_interval = 16 DWORDs (64 B, one tile row)
              | (3u << 25));      // pad_amount   = 4 DWORDs (16 B) -> 80 B row stride
  g1[1] = (int)(((unsigned)K & 0xFFFFu) << 16);                       // tensor_dim0 lo
  g1[2] = (int)(((unsigned)K >> 16) | (((unsigned)M & 0xFFFFu) << 16)); // dim0 hi | dim1 lo
  g1[3] = (int)(((unsigned)M >> 16) | ((unsigned)KSTEP << 16));       // dim1 hi | tile_dim0 = 32
  g1[4] = TILE_M;                                                     // tile_dim1 = 128
  g1[5] = K;                                                          // tensor_dim0_stride (elements)
  g1[6] = 0;
  g1[7] = 0;
  const v4i gz4 = {0, 0, 0, 0};
  const v8i gz8 = {0, 0, 0, 0, 0, 0, 0, 0};
  const unsigned long long gbase =
      (unsigned long long)(size_t)A + (size_t)mBase * K * 2ull;
#endif

  for (int k0 = 0; k0 < K; k0 += KSTEP) {
    __syncthreads();   // protect previous-iteration LDS reads
#ifdef USE_TDM
    if (wave == 0) {
      // D# group0: count=1, lds_addr, 57-bit global addr of tile start, type=2
      unsigned long long ga = gbase + (unsigned long long)k0 * 2ull;
      v4u g0;
      g0[0] = 1u;
      g0[1] = ldsAddr;
      g0[2] = (unsigned)ga;
      g0[3] = (unsigned)((ga >> 32) & 0x01FFFFFFull) | (2u << 30);
      __builtin_amdgcn_tensor_load_to_lds(g0, g1, gz4, gz4, gz8, 0);
      __builtin_amdgcn_s_wait_tensorcnt(0);   // TENSORcnt is per-wave
    }
#else
    // Fallback: manual staging (128 rows x 32 K) = 512 16-byte chunks.
#pragma unroll
    for (int i = 0; i < 2; ++i) {
      int chunk = tid + i * 256;
      int row   = chunk >> 2;
      int col   = (chunk & 3) * 8;
      const uint4* gp =
          (const uint4*)(A + (size_t)(mBase + row) * K + k0 + col);
      *(uint4*)(aTile + row * A_LDS + col) = *gp;
    }
#endif
    __syncthreads();   // publish LDS tile to all 8 waves

    // CDNA5 prefetch of next B stripe (emits global_prefetch_b8).
    if (k0 + KSTEP < K)
      __builtin_prefetch(B + (size_t)(nBase + wn + rlane) * K + k0 + KSTEP, 0, 1);

    // A fragments (16x32 bf16): lane<16 -> M=rlane, K {0..7, 16..23};
    // lane>=16 -> K {8..15, 24..31}.
    FragBF afrag[2];
#pragma unroll
    for (int i = 0; i < 2; ++i) {
      const unsigned short* src =
          aTile + (wm + i * 16 + rlane) * A_LDS + half16 * 8;
      afrag[i].u4[0] = *(const uint4*)(src);
      afrag[i].u4[1] = *(const uint4*)(src + 16);
    }
    // B fragments (32x16 bf16): lane<16 -> N=rlane, K {0..7, 16..23}; lane>=16 -> K {8..15, 24..31}.
    FragBF bfrag[2];
#pragma unroll
    for (int j = 0; j < 2; ++j) {
      const unsigned short* src =
          B + (size_t)(nBase + wn + j * 16 + rlane) * K + k0 + half16 * 8;
      bfrag[j].u4[0] = *(const uint4*)(src);
      bfrag[j].u4[1] = *(const uint4*)(src + 16);
    }

#pragma unroll
    for (int i = 0; i < 2; ++i)
#pragma unroll
      for (int j = 0; j < 2; ++j)
        acc[i][j] = __builtin_amdgcn_wmma_f32_16x16x32_bf16(
            false, afrag[i].v, false, bfrag[j].v,
            (short)0, acc[i][j], false, false);
  }

  // Store C: VGPR r holds row (r + 8*half16) of the 16x16 tile, col = rlane.
#pragma unroll
  for (int i = 0; i < 2; ++i)
#pragma unroll
    for (int j = 0; j < 2; ++j) {
      int colC = nBase + wn + j * 16 + rlane;
#pragma unroll
      for (int r = 0; r < 8; ++r) {
        int rowC = mBase + wm + i * 16 + half16 * 8 + r;
        C[(size_t)rowC * N + colC] = acc[i][j][r];
      }
    }
}

// ---------------- depthwise causal conv (width 4) + SiLU ----------------
__global__ __launch_bounds__(256)
void conv_silu_kernel(const float* __restrict__ zxbcdt,
                      const float* __restrict__ conv_w,
                      const float* __restrict__ conv_b,
                      float* __restrict__ xBCc) {
  size_t idx = (size_t)blockIdx.x * blockDim.x + threadIdx.x;
  size_t total = (size_t)BATCH * SEQLEN * CONV_DIM;
  if (idx >= total) return;
  int c  = (int)(idx % CONV_DIM);
  int bl = (int)(idx / CONV_DIM);
  int l  = bl % SEQLEN;
  int b  = bl / SEQLEN;

  float w0 = conv_w[c * 4 + 0], w1 = conv_w[c * 4 + 1];
  float w2 = conv_w[c * 4 + 2], w3 = conv_w[c * 4 + 3];
  const float* src = zxbcdt + (size_t)b * SEQLEN * D_IN_PROJ + (size_t)D_SSM + c;

  float acc = conv_b[c];
  if (l - 3 >= 0) acc += w0 * src[(size_t)(l - 3) * D_IN_PROJ];
  if (l - 2 >= 0) acc += w1 * src[(size_t)(l - 2) * D_IN_PROJ];
  if (l - 1 >= 0) acc += w2 * src[(size_t)(l - 1) * D_IN_PROJ];
  acc += w3 * src[(size_t)l * D_IN_PROJ];

  xBCc[(size_t)bl * CONV_DIM + c] = acc * sigmoidf_fast(acc);
}

// ---------------- sequential SSM scan ----------------
// One block per (batch, head). 512 threads: thread t owns p = t>>3 (headdim),
// n-range (t&7)*16 .. +15 (state dim). 16 state floats per thread in registers.
__global__ __launch_bounds__(512)
void ssm_scan_kernel(const float* __restrict__ zxbcdt,
                     const float* __restrict__ xBCc,
                     const float* __restrict__ dt_bias,
                     const float* __restrict__ A_log,
                     const float* __restrict__ Dp,
                     float* __restrict__ yraw) {
  const int b = blockIdx.x >> 6;
  const int h = blockIdx.x & 63;
  const int tid = threadIdx.x;
  const int p  = tid >> 3;
  const int nb = (tid & 7) << 4;

  __shared__ float sB[D_STATE];
  __shared__ float sC[D_STATE];
  __shared__ float sX[HEADDIM];
  __shared__ float sDt;

  const float A_h = -__expf(A_log[h]);
  const float bias = dt_bias[h];
  const float D_h = Dp[h];

  float s[16];
#pragma unroll
  for (int i = 0; i < 16; ++i) s[i] = 0.0f;

  for (int t = 0; t < SEQLEN; ++t) {
    size_t rowBC = ((size_t)b * SEQLEN + t) * CONV_DIM;
    if (tid < 128)      sB[tid]       = xBCc[rowBC + D_SSM + tid];
    else if (tid < 256) sC[tid - 128] = xBCc[rowBC + D_SSM + D_STATE + (tid - 128)];
    else if (tid < 320) sX[tid - 256] = xBCc[rowBC + h * HEADDIM + (tid - 256)];
    else if (tid == 320)
      sDt = zxbcdt[((size_t)b * SEQLEN + t) * D_IN_PROJ + (D_SSM + CONV_DIM) + h];
    __syncthreads();

    float dtr  = sDt + bias;
    float dt_s = (dtr > 20.0f) ? dtr : log1pf(__expf(dtr));   // softplus
    float dA   = __expf(dt_s * A_h);
    float xp   = sX[p];
    float cf   = xp * dt_s;

    float accy = 0.0f;
#pragma unroll
    for (int i = 0; i < 16; ++i) {
      s[i] = fmaf(dA, s[i], cf * sB[nb + i]);
      accy = fmaf(sC[nb + i], s[i], accy);
    }
    // reduce over the 8 lanes that share p (wave32 xor shuffles)
    accy += __shfl_xor(accy, 1);
    accy += __shfl_xor(accy, 2);
    accy += __shfl_xor(accy, 4);
    if ((tid & 7) == 0)
      yraw[((size_t)b * SEQLEN + t) * D_SSM + h * HEADDIM + p] = accy + D_h * xp;
    __syncthreads();
  }
}

// ---------------- gate (SiLU(z)) + RMS norm -> bf16 ----------------
__global__ __launch_bounds__(256)
void gate_norm_kernel(const float* __restrict__ zxbcdt,
                      const float* __restrict__ yraw,
                      const float* __restrict__ norm_w,
                      unsigned short* __restrict__ yn) {
  const int row = blockIdx.x;                 // b*SEQLEN + l
  const int tid = threadIdx.x;
  const float* zrow = zxbcdt + (size_t)row * D_IN_PROJ;
  const float* yrow = yraw + (size_t)row * D_SSM;

  float yg[16];
  float ss = 0.0f;
#pragma unroll
  for (int i = 0; i < 16; ++i) {
    int c = tid + i * 256;
    float z = zrow[c];
    float v = yrow[c] * (z * sigmoidf_fast(z));
    yg[i] = v;
    ss += v * v;
  }
  ss += __shfl_xor(ss, 16);
  ss += __shfl_xor(ss, 8);
  ss += __shfl_xor(ss, 4);
  ss += __shfl_xor(ss, 2);
  ss += __shfl_xor(ss, 1);

  __shared__ float red[8];
  __shared__ float rsAll;
  if ((tid & 31) == 0) red[tid >> 5] = ss;
  __syncthreads();
  if (tid == 0) {
    float a = 0.0f;
#pragma unroll
    for (int w = 0; w < 8; ++w) a += red[w];
    rsAll = rsqrtf(a / (float)D_SSM + EPS);
  }
  __syncthreads();
  float rs = rsAll;
#pragma unroll
  for (int i = 0; i < 16; ++i) {
    int c = tid + i * 256;
    yn[(size_t)row * D_SSM + c] = f32_to_bf16(yg[i] * rs * norm_w[c]);
  }
}

// ---------------- host orchestration ----------------
extern "C" void kernel_launch(void* const* d_in, const int* in_sizes, int n_in,
                              void* d_out, int out_size, void* d_ws, size_t ws_size,
                              hipStream_t stream) {
  (void)in_sizes; (void)n_in; (void)out_size; (void)ws_size;
  const float* u       = (const float*)d_in[0];
  const float* W_in    = (const float*)d_in[1];
  const float* conv_w  = (const float*)d_in[2];
  const float* conv_b  = (const float*)d_in[3];
  const float* dt_bias = (const float*)d_in[4];
  const float* A_log   = (const float*)d_in[5];
  const float* Dp      = (const float*)d_in[6];
  const float* norm_w  = (const float*)d_in[7];
  const float* W_out   = (const float*)d_in[8];
  float* out = (float*)d_out;

  const size_t n_u    = (size_t)BATCH * SEQLEN * D_MODEL;     //  8,388,608
  const size_t n_Win  = (size_t)D_IN_PROJ * D_MODEL;          // 17,432,576
  const size_t n_Wout = (size_t)D_MODEL * D_INNER;            //  8,388,608
  const size_t n_zx   = (size_t)BATCH * SEQLEN * D_IN_PROJ;   // 34,865,152
  const size_t n_xbc  = (size_t)BATCH * SEQLEN * CONV_DIM;    // 17,825,792
  const size_t n_y    = (size_t)BATCH * SEQLEN * D_SSM;       // 16,777,216

  char* ws = (char*)d_ws;
  size_t off = 0;
  auto take = [&](size_t bytes) -> void* {
    off = (off + 255) & ~(size_t)255;
    void* p = ws + off;
    off += bytes;
    return p;
  };
  unsigned short* u_bf    = (unsigned short*)take(n_u * 2);
  unsigned short* Win_bf  = (unsigned short*)take(n_Win * 2);
  unsigned short* Wout_bf = (unsigned short*)take(n_Wout * 2);
  float*          zxbcdt  = (float*)take(n_zx * 4);
  float*          xBCc    = (float*)take(n_xbc * 4);
  float*          yraw    = (float*)take(n_y * 4);
  unsigned short* yn_bf   = (unsigned short*)take(n_y * 2);

  // 1) precision conversion of GEMM operands
  cvt_bf16_kernel<<<4096, 256, 0, stream>>>(u, u_bf, n_u);
  cvt_bf16_kernel<<<4096, 256, 0, stream>>>(W_in, Win_bf, n_Win);
  cvt_bf16_kernel<<<4096, 256, 0, stream>>>(W_out, Wout_bf, n_Wout);

  // 2) GEMM1: zxbcdt(4096 x 8512) = u(4096 x 2048) * W_in^T
  {
    dim3 grid(D_IN_PROJ / TILE_N, (BATCH * SEQLEN) / TILE_M);
    wmma_gemm_bf16_kernel<<<grid, 256, 0, stream>>>(
        u_bf, Win_bf, zxbcdt, BATCH * SEQLEN, D_IN_PROJ, D_MODEL);
  }

  // 3) causal depthwise conv + SiLU on xBC slice
  {
    size_t total = (size_t)BATCH * SEQLEN * CONV_DIM;
    conv_silu_kernel<<<(unsigned)((total + 255) / 256), 256, 0, stream>>>(
        zxbcdt, conv_w, conv_b, xBCc);
  }

  // 4) sequential SSM scan: one block per (batch, head)
  ssm_scan_kernel<<<BATCH * NHEADS, 512, 0, stream>>>(
      zxbcdt, xBCc, dt_bias, A_log, Dp, yraw);

  // 5) gate + RMS norm -> bf16
  gate_norm_kernel<<<BATCH * SEQLEN, 256, 0, stream>>>(
      zxbcdt, yraw, norm_w, yn_bf);

  // 6) GEMM2: out(4096 x 2048) = yn(4096 x 4096) * W_out^T
  {
    dim3 grid(D_MODEL / TILE_N, (BATCH * SEQLEN) / TILE_M);
    wmma_gemm_bf16_kernel<<<grid, 256, 0, stream>>>(
        yn_bf, Wout_bf, out, BATCH * SEQLEN, D_MODEL, D_INNER);
  }
}
